// SelfAttention_66451734003890
// MI455X (gfx1250) — matrix operand order repaired and checked by reading, not compiled, not run on hardware
//
#include <hip/hip_runtime.h>
#include <math.h>

// Causal self-attention forward, FlashAttention-style for CDNA5 / gfx1250.
// B=4, S=2048, H=16, D=64, fp32 in/out; V_WMMA_F32_16X16X32_F16, f32 accum.
// 256 threads / 8 waves / 128 query rows per block (halves K/V HBM re-reads
// vs 64-row blocks), exp2-domain online softmax with half-wave running max,
// deferred per-lane row-sum reduction, DPP-only cross-lane reductions,
// double-buffered K/V LDS with register-staged global loads.

typedef _Float16 v16h __attribute__((ext_vector_type(16)));
typedef _Float16 v8h  __attribute__((ext_vector_type(8)));
typedef _Float16 v2h  __attribute__((ext_vector_type(2)));
typedef float    v8f  __attribute__((ext_vector_type(8)));
typedef float    v4f  __attribute__((ext_vector_type(4)));

#define B_ 4
#define S_ 2048
#define H_ 16
#define D_ 64
#define QROWS 128              // query rows per block (8 waves x 16)
// fold 1/sqrt(64) * log2(e) into Q: scores land directly in exp2 domain
#define QSCALE (0.125f * 1.44269504088896340736f)

// 16-bit A-matrix (16x32) fragment layout, CDNA5 ISA 7.12.2:
// half slot h -> VGPR r=h/2, pos=h%2 ; K = g*8 + (r<4 ? 2r+pos : 16+2(r-4)+pos)
__device__ __forceinline__ int a_kidx(int g, int h) {
    int r = h >> 1, pos = h & 1;
    int base = (r < 4) ? (2 * r + pos) : (16 + 2 * (r - 4) + pos);
    return g * 8 + base;
}

// ---- 16-lane all-reduce via DPP (stays within each 16-lane row) ----
template <int CTRL>
__device__ __forceinline__ float dppmov(float x) {
    return __int_as_float(
        __builtin_amdgcn_mov_dpp(__float_as_int(x), CTRL, 0xF, 0xF, true));
}
__device__ __forceinline__ float rowmax16(float x) {
    x = fmaxf(x, dppmov<0xB1>(x));   // quad_perm[1,0,3,2]
    x = fmaxf(x, dppmov<0x4E>(x));   // quad_perm[2,3,0,1]
    x = fmaxf(x, dppmov<0x141>(x));  // row_half_mirror
    x = fmaxf(x, dppmov<0x140>(x));  // row_mirror
    return x;
}
__device__ __forceinline__ float rowsum16(float x) {
    x += dppmov<0xB1>(x);
    x += dppmov<0x4E>(x);
    x += dppmov<0x141>(x);
    x += dppmov<0x140>(x);
    return x;
}

__global__ __launch_bounds__(256)
void flash_attn_causal_f16wmma(const float* __restrict__ qkv,
                               float* __restrict__ out) {
    __shared__ __align__(16) _Float16 Ks[2][32][D_ + 8];  // [buf][key][d]
    __shared__ __align__(16) _Float16 Vt[2][D_][32 + 8];  // [buf][d][key]
    __shared__ __align__(16) _Float16 Ps[8][16][32 + 8];  // [wave][row][key]

    const int tid  = threadIdx.x;
    const int lane = tid & 31;
    const int wave = tid >> 5;
    const int g    = lane >> 4;
    const int ln   = lane & 15;

    const int nqb   = S_ / QROWS;
    const int bh    = blockIdx.x / nqb;
    const int qb    = blockIdx.x % nqb;
    const int b     = bh / H_;
    const int h     = bh % H_;
    const int qbase = qb * QROWS;

    const int nkt = (qbase + QROWS) / 32;  // uniform causal trip count per block

    // K staging: 8 consecutive floats/thread
    const int skey = tid >> 3;
    const int sd0  = (tid & 7) * 8;
    const float* kbase = qkv + ((((long)b * S_ + skey) * 3 + 1) * H_ + h) * D_ + sd0;
    // V staging: 2 keys x 4 dims/thread (b128 loads, vectorized transposed stores)
    const int vk0 = (tid & 15) * 2;
    const int vd0 = (tid >> 4) * 4;
    const float* vbase = qkv + ((((long)b * S_ + vk0) * 3 + 2) * H_ + h) * D_ + vd0;
    const long keystep = 3L * H_ * D_;   // +1 key
    const long tstride = 32L * keystep;  // +1 tile

    // ---- Q fragments (two 16x32 A-frags), pre-scaled into exp2 domain
    const int qrow = qbase + wave * 16 + ln;
    v16h qa0, qa1;
    {
        const float* qp = qkv + ((((long)b * S_ + qrow) * 3 + 0) * H_ + h) * D_;
        #pragma unroll
        for (int hh = 0; hh < 16; ++hh) {
            int k = g * 16 + hh;
            qa0[hh] = (_Float16)(qp[k]      * QSCALE);
            qa1[hh] = (_Float16)(qp[32 + k] * QSCALE);
        }
    }

    // softmax state: one running max per wave-half; per-lane row-sum partials
    float m = -3.0e38f;
    float lsum[8];
    v8f acc[4];
    #pragma unroll
    for (int r = 0; r < 8; ++r) lsum[r] = 0.0f;
    #pragma unroll
    for (int n = 0; n < 4; ++n) { v8f z{}; acc[n] = z; }

    // ---- staging helper: K row slice (v8h store) + V 2x4 block (v2h stores)
    auto stage = [&](int buf, const float* kp, const v4f* vld) {
        v8h kk;
        #pragma unroll
        for (int d = 0; d < 8; ++d) kk[d] = (_Float16)kp[d];
        *reinterpret_cast<v8h*>(&Ks[buf][skey][sd0]) = kk;
        #pragma unroll
        for (int dd = 0; dd < 4; ++dd) {
            v2h pk;
            pk[0] = (_Float16)vld[0][dd];
            pk[1] = (_Float16)vld[1][dd];
            *reinterpret_cast<v2h*>(&Vt[buf][vd0 + dd][vk0]) = pk;
        }
    };

    // ---- prologue: stage tile 0 into buffer 0
    {
        float kreg[8];
        v4f   vreg[2];
        #pragma unroll
        for (int d = 0; d < 8; ++d) kreg[d] = kbase[d];
        #pragma unroll
        for (int j = 0; j < 2; ++j)
            vreg[j] = *reinterpret_cast<const v4f*>(vbase + j * keystep);
        stage(0, kreg, vreg);
    }
    __syncthreads();

    for (int kt = 0; kt < nkt; ++kt) {
        const int  cur       = kt & 1;
        const bool have_next = (kt + 1 < nkt);

        // ---- issue next tile's global loads early (overlap with WMMA)
        float kreg[8];
        v4f   vreg[2];
        if (have_next) {
            const float* kp = kbase + (long)(kt + 1) * tstride;
            const float* vp = vbase + (long)(kt + 1) * tstride;
            #pragma unroll
            for (int d = 0; d < 8; ++d) kreg[d] = kp[d];
            #pragma unroll
            for (int j = 0; j < 2; ++j)
                vreg[j] = *reinterpret_cast<const v4f*>(vp + j * keystep);
            if (kt + 2 < nkt) {  // warm L2 one tile further ahead
                __builtin_prefetch(kbase + (long)(kt + 2) * tstride, 0, 1);
                __builtin_prefetch(vbase + (long)(kt + 2) * tstride, 0, 1);
            }
        }

        // ---- K fragments first (one DS clause), then back-to-back WMMAs
        v16h kb00, kb10, kb01, kb11;  // [key group][d chunk]
        #pragma unroll
        for (int hh = 0; hh < 16; ++hh) {
            kb00[hh] = Ks[cur][ 0 + ln][ 0 + g * 16 + hh];
            kb10[hh] = Ks[cur][16 + ln][ 0 + g * 16 + hh];
            kb01[hh] = Ks[cur][ 0 + ln][32 + g * 16 + hh];
            kb11[hh] = Ks[cur][16 + ln][32 + g * 16 + hh];
        }
        v8f s0{}, s1{};
        s0 = __builtin_amdgcn_wmma_f32_16x16x32_f16(false, qa0, false, kb00,
                                                    (short)0, s0, false, false);
        s1 = __builtin_amdgcn_wmma_f32_16x16x32_f16(false, qa0, false, kb10,
                                                    (short)0, s1, false, false);
        s0 = __builtin_amdgcn_wmma_f32_16x16x32_f16(false, qa1, false, kb01,
                                                    (short)0, s0, false, false);
        s1 = __builtin_amdgcn_wmma_f32_16x16x32_f16(false, qa1, false, kb11,
                                                    (short)0, s1, false, false);

        // ---- causal mask (diagonal/out-of-range tiles only)
        const bool diag = (kt * 32 + 31) > (qbase + wave * 16);
        if (diag) {
            const int k0 = kt * 32 + ln;
            #pragma unroll
            for (int r = 0; r < 8; ++r) {
                const int row = qbase + wave * 16 + r + 8 * g;
                if (k0 > row)      s0[r] = -1.0e30f;
                if (k0 + 16 > row) s1[r] = -1.0e30f;
            }
        }

        // half-wave tile max (single butterfly), uniform alpha
        float tmax = fmaxf(s0[0], s1[0]);
        #pragma unroll
        for (int r = 1; r < 8; ++r) tmax = fmaxf(tmax, fmaxf(s0[r], s1[r]));
        tmax = rowmax16(tmax);
        const float mn = fmaxf(m, tmax);
        const float a  = __builtin_amdgcn_exp2f(m - mn);
        m = mn;

        // p = exp2(s - m); per-lane row-sum partials (cross-lane reduce deferred)
        float psum[8];
        #pragma unroll
        for (int r = 0; r < 8; ++r) {
            const float p0 = __builtin_amdgcn_exp2f(s0[r] - mn);
            const float p1 = __builtin_amdgcn_exp2f(s1[r] - mn);
            psum[r] = p0 + p1;
            Ps[wave][r + 8 * g][ln]      = (_Float16)p0;
            Ps[wave][r + 8 * g][16 + ln] = (_Float16)p1;
        }

        // rescale only when the running max actually moved (m is monotone)
        if (a < 1.0f) {
            #pragma unroll
            for (int r = 0; r < 8; ++r) lsum[r] *= a;
            #pragma unroll
            for (int n = 0; n < 4; ++n)
                #pragma unroll
                for (int r = 0; r < 8; ++r)
                    acc[n][r] *= a;
        }
        #pragma unroll
        for (int r = 0; r < 8; ++r) lsum[r] += psum[r];

        // wave-private LDS round-trip: C-layout -> A-layout (cross-lane)
        asm volatile("s_wait_dscnt 0x0" ::: "memory");
        v16h pa;
        #pragma unroll
        for (int hh = 0; hh < 16; ++hh)
            pa[hh] = Ps[wave][ln][g * 16 + hh];

        // ---- V fragments first, then 4 back-to-back WMMAs: P(16x32) x V(32x64)
        v16h vb[4];
        #pragma unroll
        for (int n = 0; n < 4; ++n)
            #pragma unroll
            for (int hh = 0; hh < 16; ++hh)
                vb[n][hh] = Vt[cur][n * 16 + ln][g * 16 + hh];
        #pragma unroll
        for (int n = 0; n < 4; ++n)
            acc[n] = __builtin_amdgcn_wmma_f32_16x16x32_f16(false, pa, false, vb[n],
                                                            (short)0, acc[n], false, false);

        // ---- stage next tile into the alternate buffer
        if (have_next) stage(cur ^ 1, kreg, vreg);
        __syncthreads();
    }

    // ---- epilogue: deferred row-sum reduction, normalize, write (B, S, H*D)
    #pragma unroll
    for (int r = 0; r < 8; ++r) {
        const int   srow = qbase + wave * 16 + r + 8 * g;
        const float inv  = __builtin_amdgcn_rcpf(rowsum16(lsum[r]));
        float* op = out + ((long)b * S_ + srow) * (H_ * D_) + h * D_;
        #pragma unroll
        for (int n = 0; n < 4; ++n)
            op[n * 16 + ln] = acc[n][r] * inv;
    }
}

extern "C" void kernel_launch(void* const* d_in, const int* in_sizes, int n_in,
                              void* d_out, int out_size, void* d_ws, size_t ws_size,
                              hipStream_t stream) {
    (void)in_sizes; (void)n_in; (void)d_ws; (void)ws_size; (void)out_size;
    const float* qkv = (const float*)d_in[0];
    float* out = (float*)d_out;
    dim3 grid(B_ * H_ * (S_ / QROWS));
    dim3 block(256);
    flash_attn_causal_f16wmma<<<grid, block, 0, stream>>>(qkv, out);
}
